// HeteroGraphSSM_180388626939
// MI455X (gfx1250) — matrix operand order repaired and checked by reading, not compiled, not run on hardware
//
#include <hip/hip_runtime.h>
#include <hip/hip_bf16.h>

// ---------------- problem constants (from reference) ----------------
#define NG   20000
#define NC   5000
#define NU   1000
#define PP   3
#define HH   64
#define HLX  32
#define UHH  19
#define TT   20
#define WIN  20   // UH + 1

#define EG2G 320000
#define EG2C 80000
#define EC2G 80000
#define EC2C 80000
#define EU2G 160000
#define EU2C 40000

// d_out float offsets: xs_g [NG*T], xs_c [NC*T], hs_g [NG*T*H], hs_c [NC*T*H]
#define XSG_OFF 0
#define XSC_OFF (NG * TT)                       // 400000
#define HSG_OFF (XSC_OFF + NC * TT)             // 500000
#define HSC_OFF (HSG_OFF + (long)NG * TT * HH)  // 26100000

typedef _Float16 v16h __attribute__((ext_vector_type(16)));
typedef float    v8f  __attribute__((ext_vector_type(8)));

// ---------------- shared-memory weight staging ----------------
template <int KIN>
__device__ __forceinline__ void stage_weights(
    const float* __restrict__ W1, const float* __restrict__ b1,
    const float* __restrict__ W2, const float* __restrict__ b2,
    float* sW1, float* sB1, _Float16* sW2T, float* sB2)
{
    const int tid = threadIdx.x;
    for (int i = tid; i < KIN * 64; i += blockDim.x) sW1[i] = W1[i];
    if (tid < 64) { sB1[tid] = b1[tid]; sB2[tid] = b2[tid]; }
    // W2 is [k][n] row-major; store transposed as f16 [n][k] so B-fragment
    // K-runs are contiguous (vectorizes to ds_load_b128).
    for (int i = tid; i < 64 * 64; i += blockDim.x) {
        const int k = i >> 6, n = i & 63;
        sW2T[n * 64 + k] = (_Float16)W2[i];
    }
    __syncthreads();
}

// ---------------- second layer: 16-row tile through WMMA ----------------
// xr[KIN] = this lane's input row (row m = lane&15 of the tile).
// Layer 1 (relu(x@W1+b1)) is computed by VALU directly into the A-fragment
// register layout (ISA 7.12.2, 16-bit A 16x32); layer 2 (h1@W2+b2) runs as
// 2 (K) x 4 (N) v_wmma_f32_16x16x32_f16 with C preloaded with the bias.
template <int KIN>
__device__ __forceinline__ void mlp_tile_tail(
    const float* __restrict__ xr,
    const float* __restrict__ sW1, const float* __restrict__ sB1,
    const _Float16* __restrict__ sW2T, const float* __restrict__ sB2,
    float* __restrict__ out, int row0, int nvalid)
{
    const int lane  = threadIdx.x & 31;
    const int hi    = lane >> 4;   // 0: lanes 0-15, 1: lanes 16-31
    const int nl    = lane & 15;
    const int kbase = hi * 8;      // A-fragment K striping

    v16h a0, a1;
#pragma unroll
    for (int i = 0; i < 16; ++i) {
        const int k0 = kbase + (i < 8 ? i : i + 8);   // K in [0,32)
        const int k1 = 32 + k0;                       // K in [32,64)
        float s0 = sB1[k0];
        float s1 = sB1[k1];
#pragma unroll
        for (int j = 0; j < KIN; ++j) {
            s0 = fmaf(xr[j], sW1[j * 64 + k0], s0);
            s1 = fmaf(xr[j], sW1[j * 64 + k1], s1);
        }
        a0[i] = (_Float16)fmaxf(s0, 0.0f);
        a1[i] = (_Float16)fmaxf(s1, 0.0f);
    }

    const bool full_tile = (row0 + 16 <= nvalid);   // wave-uniform

#pragma unroll
    for (int nt = 0; nt < 4; ++nt) {
        const int n = nt * 16 + nl;    // global output column for this lane
        v16h b0, b1;
#pragma unroll
        for (int i = 0; i < 16; ++i) {
            const int kb = hi * 8 + (i < 8 ? i : i + 8); // B-fragment K striping
            b0[i] = sW2T[n * 64 + kb];
            b1[i] = sW2T[n * 64 + 32 + kb];
        }
        v8f c;
        const float cb = sB2[n];
#pragma unroll
        for (int i = 0; i < 8; ++i) c[i] = cb;
        c = __builtin_amdgcn_wmma_f32_16x16x32_f16(false, a0, false, b0,
                                                   (short)0, c, false, false);
        c = __builtin_amdgcn_wmma_f32_16x16x32_f16(false, a1, false, b1,
                                                   (short)0, c, false, false);

        // C layout: element i lives at M = i + 8*hi, N = n.
        // One 64-bit base, 8 stores at constant 256B offsets.
        float* p = out + ((long)(row0 + hi * 8) * 64 + n);
        if (full_tile) {
#pragma unroll
            for (int i = 0; i < 8; ++i) p[i * 64] = c[i];
        } else {
#pragma unroll
            for (int i = 0; i < 8; ++i)
                if (row0 + hi * 8 + i < nvalid) p[i * 64] = c[i];
        }
    }
}

// ---------------- edge gate MLP: concat(pos_s, pos_d) (6) -> 64 -> 64 ----
__global__ void edge_gate_kernel(
    const float* __restrict__ ps, const float* __restrict__ pd,
    const int* __restrict__ src, const int* __restrict__ dst, int E,
    const float* __restrict__ W1, const float* __restrict__ b1,
    const float* __restrict__ W2, const float* __restrict__ b2,
    float* __restrict__ out)
{
    __shared__ float    sW1[6 * 64];
    __shared__ float    sB1[64], sB2[64];
    __shared__ _Float16 sW2T[64 * 64];
    stage_weights<6>(W1, b1, W2, b2, sW1, sB1, sW2T, sB2);

    const int wave = threadIdx.x >> 5;
    const int tile = blockIdx.x * (blockDim.x >> 5) + wave;
    const int row0 = tile * 16;
    const int lane = threadIdx.x & 31;
    const int e    = min(row0 + (lane & 15), E - 1);
    const int s    = src[e], d = dst[e];
    float xr[6];
#pragma unroll
    for (int j = 0; j < 3; ++j) { xr[j] = ps[s * 3 + j]; xr[3 + j] = pd[d * 3 + j]; }
    mlp_tile_tail<6>(xr, sW1, sB1, sW2T, sB2, out, row0, E);
}

// ---------------- history MLP: 32 -> 64 -> 64 over NG+NC rows ----------
__global__ void hist_kernel(
    const float* __restrict__ hxg, const float* __restrict__ hxc,
    const float* __restrict__ W1, const float* __restrict__ b1,
    const float* __restrict__ W2, const float* __restrict__ b2,
    float* __restrict__ out)
{
    __shared__ float    sW1[32 * 64];
    __shared__ float    sB1[64], sB2[64];
    __shared__ _Float16 sW2T[64 * 64];
    stage_weights<32>(W1, b1, W2, b2, sW1, sB1, sW2T, sB2);

    const int wave = threadIdx.x >> 5;
    const int tile = blockIdx.x * (blockDim.x >> 5) + wave;
    const int row0 = tile * 16;
    const int lane = threadIdx.x & 31;
    const int r    = min(row0 + (lane & 15), NG + NC - 1);
    const float* xrow = (r < NG) ? (hxg + (long)r * HLX)
                                 : (hxc + (long)(r - NG) * HLX);
    float xr[32];
#pragma unroll
    for (int j = 0; j < 32; ++j) xr[j] = xrow[j];
    mlp_tile_tail<32>(xr, sW1, sB1, sW2T, sB2, out, row0, NG + NC);
}

// ---------------- control MLP: window(20) -> 64 -> 64 for all T steps ----
__global__ void c2h_kernel(
    const float* __restrict__ hu, const float* __restrict__ us,
    const float* __restrict__ W1, const float* __restrict__ b1,
    const float* __restrict__ W2, const float* __restrict__ b2,
    float* __restrict__ cu_all)
{
    __shared__ float    sW1[20 * 64];
    __shared__ float    sB1[64], sB2[64];
    __shared__ _Float16 sW2T[64 * 64];
    stage_weights<20>(W1, b1, W2, b2, sW1, sB1, sW2T, sB2);

    const int t    = blockIdx.y;
    const int wave = threadIdx.x >> 5;
    const int tile = blockIdx.x * (blockDim.x >> 5) + wave;
    const int row0 = tile * 16;
    const int lane = threadIdx.x & 31;
    const int r    = min(row0 + (lane & 15), NU - 1);
    float xr[20];
#pragma unroll
    for (int j = 0; j < 20; ++j) {
        const int col = t + j;   // u_full = [history_u(19) | us(20)]
        xr[j] = (col < UHH) ? hu[r * UHH + col] : us[r * TT + (col - UHH)];
    }
    mlp_tile_tail<20>(xr, sW1, sB1, sW2T, sB2,
                      cu_all + (long)t * NU * 64, row0, NU);
}

// ---------------- zero the step-t slices of hs_g / hs_c -----------------
__global__ void zero_step_kernel(float* __restrict__ hsg,
                                 float* __restrict__ hsc, int t)
{
    long i = (long)blockIdx.x * blockDim.x + threadIdx.x;
    const long tot = (long)(NG + NC) * 64;
    if (i >= tot) return;
    const long node = i >> 6;
    const int  c    = (int)(i & 63);
    if (node < NG) hsg[(node * TT + t) * 64 + c] = 0.0f;
    else           hsc[((node - NG) * TT + t) * 64 + c] = 0.0f;
}

// ---------------- gather * gate -> scatter-add (native f32 atomics) -----
__global__ void scatter_kernel(
    const float* __restrict__ gate, const int* __restrict__ src,
    const int* __restrict__ dst, int E,
    const float* __restrict__ hsrc, long sstride,
    float* __restrict__ hdst, long dstride)
{
    long i = (long)blockIdx.x * blockDim.x + threadIdx.x;
    if (i >= (long)E * 64) return;
    const int e = (int)(i >> 6), c = (int)(i & 63);
    const float v = gate[i] * hsrc[(long)src[e] * sstride + c];
    unsafeAtomicAdd(&hdst[(long)dst[e] * dstride + c], v);  // global_atomic_add_f32
}

// ---------------- final projection hs @ h2x_W + b -----------------------
__global__ void xs_kernel(
    const float* __restrict__ hsg, const float* __restrict__ hsc,
    const float* __restrict__ w, const float* __restrict__ b,
    float* __restrict__ xsg, float* __restrict__ xsc)
{
    __shared__ float sw[64];
    if (threadIdx.x < 64) sw[threadIdx.x] = w[threadIdx.x];
    __syncthreads();
    long i = (long)blockIdx.x * blockDim.x + threadIdx.x;
    if (i >= (long)(NG + NC) * TT) return;
    const bool isG = (i < (long)NG * TT);
    const float* h = isG ? (hsg + i * 64) : (hsc + (i - (long)NG * TT) * 64);
    float acc = b[0];
#pragma unroll
    for (int c = 0; c < 64; ++c) acc = fmaf(h[c], sw[c], acc);
    if (isG) xsg[i] = acc;
    else     xsc[i - (long)NG * TT] = acc;
}

// ---------------- host side ---------------------------------------------
static inline int tile_blocks(long rows) {
    const long tiles = (rows + 15) / 16;
    return (int)((tiles + 7) / 8);   // 8 waves (tiles) per 256-thread block
}

extern "C" void kernel_launch(void* const* d_in, const int* in_sizes, int n_in,
                              void* d_out, int out_size, void* d_ws, size_t ws_size,
                              hipStream_t stream) {
    const float* pos_g = (const float*)d_in[0];
    const float* pos_c = (const float*)d_in[1];
    const float* pos_u = (const float*)d_in[2];
    const float* hxg   = (const float*)d_in[3];
    const float* hxc   = (const float*)d_in[4];
    const float* hu    = (const float*)d_in[5];
    const float* us    = (const float*)d_in[6];
    const int* s_g2g = (const int*)d_in[7];
    const int* d_g2g = (const int*)d_in[8];
    const int* s_g2c = (const int*)d_in[9];
    const int* d_g2c = (const int*)d_in[10];
    const int* s_c2g = (const int*)d_in[11];
    const int* d_c2g = (const int*)d_in[12];
    const int* s_c2c = (const int*)d_in[13];
    const int* d_c2c = (const int*)d_in[14];
    const int* s_u2g = (const int*)d_in[15];
    const int* d_u2g = (const int*)d_in[16];
    const int* s_u2c = (const int*)d_in[17];
    const int* d_u2c = (const int*)d_in[18];
    const float* aW1 = (const float*)d_in[19];
    const float* ab1 = (const float*)d_in[20];
    const float* aW2 = (const float*)d_in[21];
    const float* ab2 = (const float*)d_in[22];
    const float* bW1 = (const float*)d_in[23];
    const float* bb1 = (const float*)d_in[24];
    const float* bW2 = (const float*)d_in[25];
    const float* bb2 = (const float*)d_in[26];
    const float* hW1 = (const float*)d_in[27];
    const float* hb1 = (const float*)d_in[28];
    const float* hW2 = (const float*)d_in[29];
    const float* hb2 = (const float*)d_in[30];
    const float* cW1 = (const float*)d_in[31];
    const float* cb1 = (const float*)d_in[32];
    const float* cW2 = (const float*)d_in[33];
    const float* cb2 = (const float*)d_in[34];
    const float* xW  = (const float*)d_in[35];
    const float* xb  = (const float*)d_in[36];

    float* out = (float*)d_out;
    float* hsg = out + HSG_OFF;
    float* hsc = out + HSC_OFF;

    // workspace layout (floats)
    float* ws     = (float*)d_ws;
    float* g_g2g  = ws;
    float* g_g2c  = g_g2g + (long)EG2G * 64;
    float* g_c2g  = g_g2c + (long)EG2C * 64;
    float* g_c2c  = g_c2g + (long)EC2G * 64;
    float* g_u2g  = g_c2c + (long)EC2C * 64;
    float* g_u2c  = g_u2g + (long)EU2G * 64;
    float* h0     = g_u2c + (long)EU2C * 64;        // (NG+NC)*64
    float* cu_all = h0 + (long)(NG + NC) * 64;      // T*NU*64

    const dim3 blk(256);

    // 1) time-invariant edge gates (WMMA second layer)
    edge_gate_kernel<<<tile_blocks(EG2G), blk, 0, stream>>>(
        pos_g, pos_g, s_g2g, d_g2g, EG2G, aW1, ab1, aW2, ab2, g_g2g);
    edge_gate_kernel<<<tile_blocks(EG2C), blk, 0, stream>>>(
        pos_g, pos_c, s_g2c, d_g2c, EG2C, aW1, ab1, aW2, ab2, g_g2c);
    edge_gate_kernel<<<tile_blocks(EC2G), blk, 0, stream>>>(
        pos_c, pos_g, s_c2g, d_c2g, EC2G, aW1, ab1, aW2, ab2, g_c2g);
    edge_gate_kernel<<<tile_blocks(EC2C), blk, 0, stream>>>(
        pos_c, pos_c, s_c2c, d_c2c, EC2C, aW1, ab1, aW2, ab2, g_c2c);
    edge_gate_kernel<<<tile_blocks(EU2G), blk, 0, stream>>>(
        pos_u, pos_g, s_u2g, d_u2g, EU2G, bW1, bb1, bW2, bb2, g_u2g);
    edge_gate_kernel<<<tile_blocks(EU2C), blk, 0, stream>>>(
        pos_u, pos_c, s_u2c, d_u2c, EU2C, bW1, bb1, bW2, bb2, g_u2c);

    // 2) h0 and all control features cu[t]
    hist_kernel<<<tile_blocks(NG + NC), blk, 0, stream>>>(
        hxg, hxc, hW1, hb1, hW2, hb2, h0);
    c2h_kernel<<<dim3(tile_blocks(NU), TT), blk, 0, stream>>>(
        hu, us, cW1, cb1, cW2, cb2, cu_all);

    // 3) scan: zero step slice, scatter control, scatter graph messages
    const long ZN = (long)(NG + NC) * 64;
    const int  zb = (int)((ZN + 255) / 256);
    for (int t = 0; t < TT; ++t) {
        zero_step_kernel<<<zb, blk, 0, stream>>>(hsg, hsc, t);

        scatter_kernel<<<(int)(((long)EU2G * 64 + 255) / 256), blk, 0, stream>>>(
            g_u2g, s_u2g, d_u2g, EU2G, cu_all + (long)t * NU * 64, 64,
            hsg + t * 64, (long)TT * 64);
        scatter_kernel<<<(int)(((long)EU2C * 64 + 255) / 256), blk, 0, stream>>>(
            g_u2c, s_u2c, d_u2c, EU2C, cu_all + (long)t * NU * 64, 64,
            hsc + t * 64, (long)TT * 64);

        const float* srcg = (t == 0) ? h0 : (hsg + (t - 1) * 64);
        const float* srcc = (t == 0) ? (h0 + (long)NG * 64) : (hsc + (t - 1) * 64);
        const long   ssg  = (t == 0) ? 64 : (long)TT * 64;
        const long   ssc  = (t == 0) ? 64 : (long)TT * 64;

        scatter_kernel<<<(int)(((long)EG2G * 64 + 255) / 256), blk, 0, stream>>>(
            g_g2g, s_g2g, d_g2g, EG2G, srcg, ssg, hsg + t * 64, (long)TT * 64);
        scatter_kernel<<<(int)(((long)EC2G * 64 + 255) / 256), blk, 0, stream>>>(
            g_c2g, s_c2g, d_c2g, EC2G, srcc, ssc, hsg + t * 64, (long)TT * 64);
        scatter_kernel<<<(int)(((long)EG2C * 64 + 255) / 256), blk, 0, stream>>>(
            g_g2c, s_g2c, d_g2c, EG2C, srcg, ssg, hsc + t * 64, (long)TT * 64);
        scatter_kernel<<<(int)(((long)EC2C * 64 + 255) / 256), blk, 0, stream>>>(
            g_c2c, s_c2c, d_c2c, EC2C, srcc, ssc, hsc + t * 64, (long)TT * 64);
    }

    // 4) xs projection
    const long XN = (long)(NG + NC) * TT;
    xs_kernel<<<(int)((XN + 255) / 256), blk, 0, stream>>>(
        hsg, hsc, xW, xb, out + XSG_OFF, out + XSC_OFF);
}
